// Attn_18880676233761
// MI455X (gfx1250) — matrix operand order repaired and checked by reading, not compiled, not run on hardware
//
#include <hip/hip_runtime.h>

typedef __attribute__((ext_vector_type(16))) __bf16 v16bf;
typedef __attribute__((ext_vector_type(8)))  __bf16 v8bf;
typedef __attribute__((ext_vector_type(8)))  float  v8f;

#define DEV __device__ __forceinline__

static constexpr int B_ = 4, T_ = 2048, D_ = 512, NH_ = 8, NKV_ = 4, HD_ = 64;

// ---------------- WMMA helpers ----------------

DEV v8f wmma_bf16(v16bf a, v16bf b, v8f c) {
  // D = A(16x32 bf16) * B(32x16 bf16) + C(16x16 f32)
  return __builtin_amdgcn_wmma_f32_16x16x32_bf16(
      /*neg_a=*/false, a, /*neg_b=*/false, b,
      /*c_mod=*/(short)0, c, /*reuse_a=*/false, /*reuse_b=*/false);
}

// A-matrix 16x32 (16-bit): lane<16 -> M=lane, K={k0..k0+7, k0+16..k0+23}
//                          lane>=16 -> M=lane-16, K={k0+8..k0+15, k0+24..k0+31}
DEV v16bf load_a_frag(const __bf16* base, int ld, int k0) {
  int lane = threadIdx.x & 31;
  const __bf16* p = base + (size_t)(lane & 15) * ld + k0 + ((lane >> 4) << 3);
  v8bf lo = *(const v8bf*)p;
  v8bf hi = *(const v8bf*)(p + 16);
  return __builtin_shufflevector(lo, hi, 0,1,2,3,4,5,6,7,8,9,10,11,12,13,14,15);
}

// B-matrix 32x16 (16-bit): lane<16 -> N=lane, K={k0..k0+7, k0+16..k0+23}; lane>=16 -> K={k0+8..k0+15, k0+24..k0+31}.
// "base" is row-major with B's column n == row n of base (K contiguous).
DEV v16bf load_b_frag(const __bf16* base, int ld, int k0) {
  int lane = threadIdx.x & 31;
  const __bf16* p = base + (size_t)(lane & 15) * ld + k0 + ((lane >> 4) << 3);
  v8bf lo = *(const v8bf*)p;
  v8bf hi = *(const v8bf*)(p + 16);
  return __builtin_shufflevector(lo, hi, 0,1,2,3,4,5,6,7,8,9,10,11,12,13,14,15);
}

// Cross-lane reductions within 16-lane halves (matches C-fragment row striping).
DEV float half_sum(float v) {
#pragma unroll
  for (int m = 1; m < 16; m <<= 1) v += __shfl_xor(v, m, 32);
  return v;
}
DEV float half_max(float v) {
#pragma unroll
  for (int m = 1; m < 16; m <<= 1) v = fmaxf(v, __shfl_xor(v, m, 32));
  return v;
}

// LDS offset (bytes) of a generic pointer to LDS: low 32 bits of the flat address.
DEV unsigned lds_off(const void* p) { return (unsigned)(unsigned long long)p; }

// One lane-cooperative async 16B copy: global -> LDS, tracked by ASYNCcnt.
DEV void async_copy_b128(unsigned lds_byte, unsigned long long gaddr) {
  asm volatile("global_load_async_to_lds_b128 %0, %1, off"
               :: "v"(lds_byte), "v"(gaddr) : "memory");
}

// ---------------- f32 -> bf16 conversion ----------------

__global__ void cvt_bf16(const float* __restrict__ s, __bf16* __restrict__ d, int n) {
  int i = blockIdx.x * blockDim.x + threadIdx.x;
  if (i < n) d[i] = (__bf16)s[i];
}

// ---------------- fused QKV projection + norm + rope + gates ----------------
// grid: (T/16, NH + 2*NKV, B), block: 32 (one wave owns a 16(T) x 64(head) tile)

__global__ __launch_bounds__(32) void qkv_fused(
    const __bf16* __restrict__ xb,
    const __bf16* __restrict__ Wqb,
    const __bf16* __restrict__ Wkb,
    const __bf16* __restrict__ Wvb,
    const float*  __restrict__ qg,
    const float*  __restrict__ vres_gate,
    const float*  __restrict__ v_first,
    __bf16* __restrict__ qb,
    __bf16* __restrict__ kb,
    __bf16* __restrict__ vbT,
    float*  __restrict__ vout)
{
  const int tt   = blockIdx.x;           // 16-row tile of T
  const int slot = blockIdx.y;           // 0..7 = q heads, 8..11 = k heads, 12..15 = v heads
  const int b    = blockIdx.z;
  const int lane = threadIdx.x & 31;
  const int c    = lane & 15;            // column within 16-wide tile
  const int hi   = (lane >> 4) << 3;     // +8 rows for upper half-wave

  const __bf16* W;
  int obase;
  if (slot < 8)       { W = Wqb; obase = slot * HD_; }
  else if (slot < 12) { W = Wkb; obase = (slot - 8) * HD_; }
  else                { W = Wvb; obase = (slot - 12) * HD_; }

  const __bf16* xrow = xb + ((size_t)b * T_ + tt * 16) * D_;

  v8f acc[4] = {};
  for (int k0 = 0; k0 < D_; k0 += 32) {
    if (k0 + 64 < D_)
      __builtin_prefetch((const void*)(xrow + (size_t)c * D_ + k0 + 64), 0, 3);
    v16bf a = load_a_frag(xrow, D_, k0);
#pragma unroll
    for (int ct = 0; ct < 4; ct++) {
      v16bf bm = load_b_frag(W + (size_t)(obase + ct * 16) * D_, D_, k0);
      acc[ct] = wmma_bf16(a, bm, acc[ct]);
    }
  }

  if (slot < 12) {
    // RMSNorm over the 64 head columns (per q/k row), entirely in registers.
#pragma unroll
    for (int r = 0; r < 8; r++) {
      float s = 0.f;
#pragma unroll
      for (int ct = 0; ct < 4; ct++) s += acc[ct][r] * acc[ct][r];
      s = half_sum(s);
      float f = rsqrtf(s * (1.0f / 64.0f) + 1.1920929e-07f);
#pragma unroll
      for (int ct = 0; ct < 4; ct++) acc[ct][r] *= f;
    }
    // Partial RoPE on columns 0..7 (tile 0); pair (c, c^4) via lane shuffle.
#pragma unroll
    for (int r = 0; r < 8; r++) {
      float partner = __shfl_xor(acc[0][r], 4, 32);
      if (c < 8) {
        int j = c & 3;
        float invf = __expf(-(float)j * 2.3025850929940457f); // 10000^(-j/4)
        float ang  = (float)(tt * 16 + hi + r) * invf;
        float sn, cs;
        __sincosf(ang, &sn, &cs);
        acc[0][r] = (c < 4) ? ( acc[0][r] * cs + partner * sn)
                            : (-acc[0][r] * cs + partner * sn);
      }
    }
  }

  if (slot < 8) {
    // q: apply head gain and fold in 1/sqrt(hd); store bf16 (B,NH,T,hd)
    float scale = qg[slot] * 0.125f;
    __bf16* qd = qb + (((size_t)b * NH_ + slot) * T_ + tt * 16 + hi) * HD_;
#pragma unroll
    for (int ct = 0; ct < 4; ct++)
#pragma unroll
      for (int r = 0; r < 8; r++)
        qd[(size_t)r * HD_ + ct * 16 + c] = (__bf16)(acc[ct][r] * scale);
  } else if (slot < 12) {
    __bf16* kd = kb + (((size_t)b * NKV_ + (slot - 8)) * T_ + tt * 16 + hi) * HD_;
#pragma unroll
    for (int ct = 0; ct < 4; ct++)
#pragma unroll
      for (int r = 0; r < 8; r++)
        kd[(size_t)r * HD_ + ct * 16 + c] = (__bf16)acc[ct][r];
  } else {
    // v: value-residual gate mix; f32 to d_out, bf16 transposed (hd x T) for PV.
    const int h = slot - 12;
    float g = 1.0f / (1.0f + __expf(-vres_gate[0]));
    const float* vf = v_first + (((size_t)b * NKV_ + h) * T_ + tt * 16 + hi) * HD_;
    float*       vo = vout    + (((size_t)b * NKV_ + h) * T_ + tt * 16 + hi) * HD_;
    __bf16*      vt = vbT     + ((size_t)b * NKV_ + h) * HD_ * T_;
#pragma unroll
    for (int ct = 0; ct < 4; ct++)
#pragma unroll
      for (int r = 0; r < 8; r++) {
        int col = ct * 16 + c;
        float vv = (1.0f - g) * acc[ct][r] + g * vf[(size_t)r * HD_ + col];
        vo[(size_t)r * HD_ + col] = vv;
        vt[(size_t)col * T_ + tt * 16 + hi + r] = (__bf16)vv;
      }
  }
}

// ---------------- flash attention with async double-buffered K/V staging ----------------
// grid: (T/16, NH, B), block: 32. One wave owns a 16-row q tile x hd=64.
// K tile: 32 keys x 64 d (4KB, contiguous).  V tile: 64 d x 32 keys from V^T (4KB, strided rows).

__global__ __launch_bounds__(32) void attn_flash(
    const __bf16* __restrict__ qb,
    const __bf16* __restrict__ kb,
    const __bf16* __restrict__ vbT,
    __bf16* __restrict__ yb)
{
  __shared__ __bf16 ldsK[2][32 * 64];   // [key][d]
  __shared__ __bf16 ldsV[2][64 * 32];   // [d][key]
  __shared__ __bf16 ldsP[16 * 32];      // P bounce: C-layout -> A-layout

  const int qt = blockIdx.x, h = blockIdx.y, b = blockIdx.z;
  const int lane = threadIdx.x & 31;
  const int c  = lane & 15;
  const int hi = (lane >> 4) << 3;
  const int hk = h >> 1;             // GQA: rep = NH/NKV = 2

  const __bf16* qh = qb  + (((size_t)b * NH_  + h ) * T_ + qt * 16) * HD_;
  const __bf16* kh = kb  + (((size_t)b * NKV_ + hk) * T_) * HD_;
  const __bf16* vT = vbT + ((size_t)b * NKV_ + hk) * HD_ * T_;

  // Stage K+V tile `ti` (keys ti*32..ti*32+31) into buffer `bi`: 16 async b128 ops.
  auto stage = [&](int ti, int bi) {
    const int k0 = ti * 32;
    // K: 4KB contiguous
    unsigned long long gk = (unsigned long long)(const void*)(kh + (size_t)k0 * HD_);
    unsigned lk = lds_off(&ldsK[bi][0]);
#pragma unroll
    for (int i = 0; i < 8; i++)
      async_copy_b128(lk + i * 512 + lane * 16, gk + i * 512 + lane * 16);
    // V^T: 64 rows (d), 64B each (32 keys); 4 lanes cover one row.
    unsigned lv = lds_off(&ldsV[bi][0]);
#pragma unroll
    for (int i = 0; i < 8; i++) {
      int row = i * 8 + (lane >> 2);
      unsigned long long gv =
          (unsigned long long)(const void*)(vT + (size_t)row * T_ + k0) + (lane & 3) * 16;
      async_copy_b128(lv + row * 64 + (lane & 3) * 16, gv);
    }
  };

  // Q fragments resident for the whole sweep (hd=64 -> two K=32 chunks)
  v16bf qa0 = load_a_frag(qh, HD_, 0);
  v16bf qa1 = load_a_frag(qh, HD_, 32);

  v8f acc[4] = {};
  float mrow[8], lrow[8];
#pragma unroll
  for (int r = 0; r < 8; r++) { mrow[r] = -3.0e38f; lrow[r] = 0.f; }

  const int kend   = qt * 16 + 16;          // causal: keys 0..qt*16+15
  const int ntiles = (kend + 31) >> 5;

  stage(0, 0);
  for (int ti = 0; ti < ntiles; ti++) {
    const int k0 = ti * 32;
    const int bi = ti & 1;

    if (ti + 1 < ntiles) {
      stage(ti + 1, bi ^ 1);
      // drain this tile's 16 ops, leave the next tile's 16 in flight
      asm volatile("s_wait_asynccnt 0x10" ::: "memory");
    } else {
      asm volatile("s_wait_asynccnt 0x0" ::: "memory");
    }

    // S = Q K^T : two 16x16 tiles from LDS K, K-dim = hd = 64
    const __bf16* kr0 = &ldsK[bi][0];
    const __bf16* kr1 = &ldsK[bi][16 * 64];
    v8f s0 = {}, s1 = {};
    s0 = wmma_bf16(qa0, load_b_frag(kr0, HD_, 0),  s0);
    s0 = wmma_bf16(qa1, load_b_frag(kr0, HD_, 32), s0);
    s1 = wmma_bf16(qa0, load_b_frag(kr1, HD_, 0),  s1);
    s1 = wmma_bf16(qa1, load_b_frag(kr1, HD_, 32), s1);

    // causal mask + online softmax in C layout
#pragma unroll
    for (int r = 0; r < 8; r++) {
      int qrow = qt * 16 + hi + r;
      if (k0 + c      > qrow) s0[r] = -3.0e38f;
      if (k0 + 16 + c > qrow) s1[r] = -3.0e38f;
      float mt   = fmaxf(half_max(s0[r]), half_max(s1[r]));
      float mnew = fmaxf(mrow[r], mt);
      float corr = __expf(mrow[r] - mnew);
      mrow[r] = mnew;
      float p0 = __expf(s0[r] - mnew);
      float p1 = __expf(s1[r] - mnew);
      lrow[r] = lrow[r] * corr + half_sum(p0) + half_sum(p1);
#pragma unroll
      for (int ct = 0; ct < 4; ct++) acc[ct][r] *= corr;
      ldsP[(size_t)(hi + r) * 32 + c]      = (__bf16)p0;
      ldsP[(size_t)(hi + r) * 32 + 16 + c] = (__bf16)p1;
    }

    // Y += P * V : A = P (LDS), B = V tile rows in LDS (32 keys each)
    v16bf pa = load_a_frag(&ldsP[0], 32, 0);
#pragma unroll
    for (int ct = 0; ct < 4; ct++) {
      v16bf vb_ = load_b_frag(&ldsV[bi][(size_t)(ct * 16) * 32], 32, 0);
      acc[ct] = wmma_bf16(pa, vb_, acc[ct]);
    }
  }

  __bf16* yd = yb + ((size_t)b * T_ + qt * 16 + hi) * D_ + h * HD_;
#pragma unroll
  for (int ct = 0; ct < 4; ct++)
#pragma unroll
    for (int r = 0; r < 8; r++)
      yd[(size_t)r * D_ + ct * 16 + c] = (__bf16)(acc[ct][r] / lrow[r]);
}

// ---------------- output projection ----------------
// grid: (B*T/16, D/64), block: 32

__global__ __launch_bounds__(32) void out_proj(
    const __bf16* __restrict__ yb,
    const __bf16* __restrict__ Wpb,
    float* __restrict__ out)
{
  const int mt = blockIdx.x;
  const int nt = blockIdx.y;
  const int lane = threadIdx.x & 31;
  const int c = lane & 15, hi = (lane >> 4) << 3;
  const __bf16* arow = yb + (size_t)mt * 16 * D_;
  v8f acc[4] = {};
  for (int k0 = 0; k0 < D_; k0 += 32) {
    if (k0 + 64 < D_)
      __builtin_prefetch((const void*)(arow + (size_t)c * D_ + k0 + 64), 0, 3);
    v16bf a = load_a_frag(arow, D_, k0);
#pragma unroll
    for (int ct = 0; ct < 4; ct++) {
      v16bf bm = load_b_frag(Wpb + (size_t)(nt * 64 + ct * 16) * D_, D_, k0);
      acc[ct] = wmma_bf16(a, bm, acc[ct]);
    }
  }
  float* od = out + ((size_t)mt * 16 + hi) * D_ + nt * 64;
#pragma unroll
  for (int ct = 0; ct < 4; ct++)
#pragma unroll
    for (int r = 0; r < 8; r++)
      od[(size_t)r * D_ + ct * 16 + c] = acc[ct][r];
}

// ---------------- launch ----------------

extern "C" void kernel_launch(void* const* d_in, const int* in_sizes, int n_in,
                              void* d_out, int out_size, void* d_ws, size_t ws_size,
                              hipStream_t stream) {
  (void)in_sizes; (void)n_in; (void)out_size; (void)ws_size;

  const float* x  = (const float*)d_in[0];
  const float* Wq = (const float*)d_in[1];
  const float* Wk = (const float*)d_in[2];
  const float* Wv = (const float*)d_in[3];
  const float* Wp = (const float*)d_in[4];
  const float* qg = (const float*)d_in[5];
  const float* vg = (const float*)d_in[6];
  const float* vf = (const float*)d_in[7];

  char* w = (char*)d_ws;
  auto carve = [&](size_t elems) {
    __bf16* p = (__bf16*)w;
    w += (elems * sizeof(__bf16) + 255) & ~(size_t)255;
    return p;
  };
  __bf16* xb  = carve((size_t)B_ * T_ * D_);
  __bf16* Wqb = carve((size_t)D_ * D_);
  __bf16* Wkb = carve((size_t)NKV_ * HD_ * D_);
  __bf16* Wvb = carve((size_t)NKV_ * HD_ * D_);
  __bf16* Wpb = carve((size_t)D_ * D_);
  __bf16* qb  = carve((size_t)B_ * NH_ * T_ * HD_);
  __bf16* kb  = carve((size_t)B_ * NKV_ * T_ * HD_);
  __bf16* vbT = carve((size_t)B_ * NKV_ * HD_ * T_);
  __bf16* yb  = carve((size_t)B_ * T_ * D_);

  float* out  = (float*)d_out;
  float* vout = out + (size_t)B_ * T_ * D_;

  auto cvt = [&](const float* s, __bf16* d, int n) {
    cvt_bf16<<<dim3((n + 255) / 256), dim3(256), 0, stream>>>(s, d, n);
  };
  cvt(x,  xb,  B_ * T_ * D_);
  cvt(Wq, Wqb, D_ * D_);
  cvt(Wk, Wkb, NKV_ * HD_ * D_);
  cvt(Wv, Wvb, NKV_ * HD_ * D_);
  cvt(Wp, Wpb, D_ * D_);

  qkv_fused<<<dim3(T_ / 16, NH_ + 2 * NKV_, B_), dim3(32), 0, stream>>>(
      xb, Wqb, Wkb, Wvb, qg, vg, vf, qb, kb, vbT, vout);
  attn_flash<<<dim3(T_ / 16, NH_, B_), dim3(32), 0, stream>>>(qb, kb, vbT, yb);
  out_proj<<<dim3(B_ * T_ / 16, D_ / 64, 1), dim3(32), 0, stream>>>(yb, Wpb, out);
}
